// PharmMatchNetFast_90237262889464
// MI455X (gfx1250) — matrix operand order repaired and checked by reading, not compiled, not run on hardware
//
#include <hip/hip_runtime.h>
#include <hip/hip_bf16.h>

typedef __attribute__((ext_vector_type(16))) _Float16 v16h;
typedef __attribute__((ext_vector_type(8)))  _Float16 v8h;
typedef __attribute__((ext_vector_type(8)))  float    v8f;

#define NB 128
#define NL 128
#define NP 1024
#define ND 128
#define NM 64
#define NEGF (-3.402823466e+38f)
#define SCALE_D 0.08838834764831845f      // 1/sqrt(128)
#define HALF_LOG_2PI 0.9189385332046727f  // 0.5*log(2*pi)

// ---------------------------------------------------------------- helpers
static __device__ __forceinline__ float gelu_f(float x) {
  return 0.5f * x * (1.0f + erff(x * 0.7071067811865475f));
}
static __device__ __forceinline__ float softplus_f(float x) {
  return (x > 20.0f) ? x : log1pf(expf(x));
}
static __device__ __forceinline__ v8f wmma16(v16h a, v16h b, v8f c) {
  return __builtin_amdgcn_wmma_f32_16x16x32_f16(false, a, false, b, (short)0, c,
                                                false, false);
}
// A fragment, f16 row-major source (LDS or global). ISA 16-bit A 16x32 layout:
// lanes 0-15: row=m0+lane, halves 0..7 = K k0..k0+7, halves 8..15 = K k0+16..k0+23
// lanes 16-31: same rows, K shifted by +8 (and +24).
static __device__ __forceinline__ v16h load_a_f16(const _Float16* A, int lda,
                                                  int m0, int k0, int lane) {
  const _Float16* p = A + (size_t)(m0 + (lane & 15)) * lda + k0 + ((lane >> 4) ? 8 : 0);
  v8h lo = *(const v8h*)p;
  v8h hi = *(const v8h*)(p + 16);
  v16h f;
#pragma unroll
  for (int i = 0; i < 8; ++i) { f[i] = lo[i]; f[i + 8] = hi[i]; }
  return f;
}
// Same fragment but converting from f32 source on the fly.
static __device__ __forceinline__ v16h load_a_f32(const float* A, int lda,
                                                  int m0, int k0, int lane) {
  const float* p = A + (size_t)(m0 + (lane & 15)) * lda + k0 + ((lane >> 4) ? 8 : 0);
  v16h f;
#pragma unroll
  for (int i = 0; i < 8; ++i) { f[i] = (_Float16)p[i]; f[i + 8] = (_Float16)p[16 + i]; }
  return f;
}
// B fragment from an N-major f16 matrix Wt[N][ldk] (logical B[k][n] = Wt[n][k]).
// lanes 0-15: col=n0+lane, K k0..k0+7, k0+16..k0+23 ; lanes 16-31: col=n0+lane-16, K k0+8..k0+15, k0+24..k0+31.
static __device__ __forceinline__ v16h load_b_nm(const _Float16* Wt, int ldk,
                                                 int n0, int k0, int lane) {
  const _Float16* p = Wt + (size_t)(n0 + (lane & 15)) * ldk + k0 + ((lane >> 4) ? 8 : 0);
  v8h lo = *(const v8h*)p;
  v8h hi = *(const v8h*)(p + 16);
  v16h f;
#pragma unroll
  for (int i = 0; i < 8; ++i) { f[i] = lo[i]; f[i + 8] = hi[i]; }
  return f;
}
// Per-wave LayerNorm of a 16x128 tile held as 8 v8f accumulators.
// Element (m = 16*tm + 8*(lane>=16) + r, n = 16*tn + (lane&15)) lives in acc[tn][r].
static __device__ __forceinline__ void layer_norm_wave(v8f* acc, const float* g,
                                                       const float* bv, int ln) {
  float gr[8], br[8];
#pragma unroll
  for (int tn = 0; tn < 8; ++tn) { gr[tn] = g[tn * 16 + ln]; br[tn] = bv[tn * 16 + ln]; }
  for (int r = 0; r < 8; ++r) {
    float s = 0.0f;
#pragma unroll
    for (int tn = 0; tn < 8; ++tn) s += acc[tn][r];
    for (int off = 8; off >= 1; off >>= 1) s += __shfl_xor(s, off, 32);
    float mean = s * (1.0f / 128.0f);
    float vsum = 0.0f;
#pragma unroll
    for (int tn = 0; tn < 8; ++tn) { float d = acc[tn][r] - mean; vsum += d * d; }
    for (int off = 8; off >= 1; off >>= 1) vsum += __shfl_xor(vsum, off, 32);
    float rstd = rsqrtf(vsum * (1.0f / 128.0f) + 1e-5f);
#pragma unroll
    for (int tn = 0; tn < 8; ++tn)
      acc[tn][r] = (acc[tn][r] - mean) * rstd * gr[tn] + br[tn];
  }
}
static __device__ __forceinline__ void top4_insert(float* v, int* ix, float x, int id) {
  if (x <= v[3]) return;
  v[3] = x; ix[3] = id;
  for (int j = 3; j > 0; --j) {
    if (v[j] > v[j - 1]) {
      float tv = v[j]; v[j] = v[j - 1]; v[j - 1] = tv;
      int   ti = ix[j]; ix[j] = ix[j - 1]; ix[j - 1] = ti;
    } else break;
  }
}
static __device__ __forceinline__ float block_sum(float v, float* red) {
  int t = threadIdx.x;
  red[t] = v; __syncthreads();
  for (int s = 128; s >= 1; s >>= 1) {
    if (t < s) red[t] += red[t + s];
    __syncthreads();
  }
  float r = red[0]; __syncthreads();
  return r;
}

// ---------------------------------------------------------------- weight prep
// f32 W[K][N] -> f16 N-major Wt[N][K]
__global__ void k_cvt_nmajor(const float* __restrict__ W, _Float16* __restrict__ O,
                             int K, int N) {
  int i = blockIdx.x * blockDim.x + threadIdx.x;
  if (i >= K * N) return;
  int n = i / K, k = i - n * K;
  O[(size_t)n * K + k] = (_Float16)W[(size_t)k * N + n];
}

// ---------------------------------------------------------------- hL chain
__global__ void __launch_bounds__(256) k_embedL(
    const float* __restrict__ lx, const int* __restrict__ ltyp,
    const float* __restrict__ le_emb, const float* __restrict__ le_w1,
    const float* __restrict__ le_b1, const _Float16* __restrict__ le_w2_t,
    const float* __restrict__ le_b2, const float* __restrict__ le_g,
    const float* __restrict__ le_lnb, const _Float16* __restrict__ projL_t,
    _Float16* __restrict__ hL16) {
  extern __shared__ char smem_raw[];
  _Float16* act = (_Float16*)smem_raw;  // [128][136] f16
  const int b = blockIdx.x;
  const int t = threadIdx.x, lane = t & 31, wave = t >> 5;
  const int ln = lane & 15, half = lane >> 4;
  __builtin_prefetch(le_w2_t, 0, 3);
  __builtin_prefetch(projL_t, 0, 3);
  // stage 1: gelu(l_x @ le_w1 + b1) -> LDS f16
  const float* lxb = lx + (size_t)b * NL * 8;
  for (int i = t; i < NL * ND; i += 256) {
    int row = i >> 7, col = i & 127;
    float v = le_b1[col];
#pragma unroll
    for (int k = 0; k < 8; ++k) v += lxb[row * 8 + k] * le_w1[k * ND + col];
    act[row * 136 + col] = (_Float16)gelu_f(v);
  }
  __syncthreads();
  const int tm = wave;
  // stage 2: t2 = act @ le_w2 + b2 + emb ; LayerNorm
  v16h af[4];
#pragma unroll
  for (int ks = 0; ks < 4; ++ks) af[ks] = load_a_f16(act, 136, tm * 16, ks * 32, lane);
  v8f acc[8];
  for (int tn = 0; tn < 8; ++tn) {
    v8f c = {};
#pragma unroll
    for (int ks = 0; ks < 4; ++ks)
      c = wmma16(af[ks], load_b_nm(le_w2_t, 128, tn * 16, ks * 32, lane), c);
    int n = tn * 16 + ln;
#pragma unroll
    for (int r = 0; r < 8; ++r) {
      int m = tm * 16 + half * 8 + r;
      int typ = ltyp[b * NL + m];
      c[r] += le_b2[n] + le_emb[(size_t)typ * ND + n];
    }
    acc[tn] = c;
  }
  layer_norm_wave(acc, le_g, le_lnb, ln);
  // write normalized back into our own rows (wave-private, no barrier needed)
  for (int tn = 0; tn < 8; ++tn) {
    int n = tn * 16 + ln;
#pragma unroll
    for (int r = 0; r < 8; ++r) {
      int m = tm * 16 + half * 8 + r;
      act[m * 136 + n] = (_Float16)acc[tn][r];
    }
  }
  // stage 3: hL = norm @ projL
#pragma unroll
  for (int ks = 0; ks < 4; ++ks) af[ks] = load_a_f16(act, 136, tm * 16, ks * 32, lane);
  _Float16* outp = hL16 + (size_t)b * NL * ND;
  for (int tn = 0; tn < 8; ++tn) {
    v8f c = {};
#pragma unroll
    for (int ks = 0; ks < 4; ++ks)
      c = wmma16(af[ks], load_b_nm(projL_t, 128, tn * 16, ks * 32, lane), c);
    int n = tn * 16 + ln;
#pragma unroll
    for (int r = 0; r < 8; ++r) {
      int m = tm * 16 + half * 8 + r;
      outp[(size_t)m * ND + n] = (_Float16)c[r];
    }
  }
}

// ---------------------------------------------------------------- hP chain + sigma
__global__ void __launch_bounds__(256) k_embedP(
    const float* __restrict__ px, const float* __restrict__ pscore,
    const float* __restrict__ prad, const int* __restrict__ ptyp,
    const float* __restrict__ pe_emb, const _Float16* __restrict__ pe_w1_t,
    const float* __restrict__ pe_b1, const _Float16* __restrict__ pe_w2_t,
    const float* __restrict__ pe_b2, const float* __restrict__ ps_w1,
    const float* __restrict__ ps_b1, const _Float16* __restrict__ ps_w2_t,
    const float* __restrict__ ps_b2, const float* __restrict__ pe_g,
    const float* __restrict__ pe_lnb, const _Float16* __restrict__ projP_t,
    const _Float16* __restrict__ sg_w1_t, const float* __restrict__ sg_b1,
    const float* __restrict__ sg_w2, const float* __restrict__ sg_b2,
    _Float16* __restrict__ hP16, float* __restrict__ p_sigma) {
  extern __shared__ char smem_raw[];
  _Float16* act  = (_Float16*)smem_raw;   // [128][136]
  _Float16* act2 = act + 128 * 136;       // [128][136]
  const int b = blockIdx.x >> 3, chunk = blockIdx.x & 7;
  const int pr0 = chunk * 128;
  const int t = threadIdx.x, lane = t & 31, wave = t >> 5;
  const int ln = lane & 15, half = lane >> 4;
  __builtin_prefetch(pe_w1_t, 0, 3);
  __builtin_prefetch(pe_w2_t, 0, 3);
  // stage 0b: score MLP hidden -> act2
  for (int i = t; i < 128 * ND; i += 256) {
    int row = i >> 7, col = i & 127;
    float sc = pscore[(size_t)b * NP + pr0 + row];
    act2[row * 136 + col] = (_Float16)gelu_f(sc * ps_w1[col] + ps_b1[col]);
  }
  __syncthreads();
  const int tm = wave;
  // stage 1: u = gelu(p_x @ pe_w1 + b1)  (A fragments straight from global f32)
  const float* pxb = px + ((size_t)b * NP + pr0 + tm * 16) * 192;
  {
    v16h af6[6];
#pragma unroll
    for (int ks = 0; ks < 6; ++ks) af6[ks] = load_a_f32(pxb, 192, 0, ks * 32, lane);
    for (int tn = 0; tn < 8; ++tn) {
      v8f c = {};
#pragma unroll
      for (int ks = 0; ks < 6; ++ks)
        c = wmma16(af6[ks], load_b_nm(pe_w1_t, 192, tn * 16, ks * 32, lane), c);
      int n = tn * 16 + ln;
#pragma unroll
      for (int r = 0; r < 8; ++r) {
        int m = tm * 16 + half * 8 + r;
        act[m * 136 + n] = (_Float16)gelu_f(c[r] + pe_b1[n]);
      }
    }
  }
  // stage 2: t = u @ pe_w2 + s @ ps_w2 + b2s + emb ; LayerNorm
  v16h a1[4], a2[4];
#pragma unroll
  for (int ks = 0; ks < 4; ++ks) {
    a1[ks] = load_a_f16(act, 136, tm * 16, ks * 32, lane);
    a2[ks] = load_a_f16(act2, 136, tm * 16, ks * 32, lane);
  }
  v8f acc[8];
  for (int tn = 0; tn < 8; ++tn) {
    v8f c = {};
#pragma unroll
    for (int ks = 0; ks < 4; ++ks)
      c = wmma16(a1[ks], load_b_nm(pe_w2_t, 128, tn * 16, ks * 32, lane), c);
#pragma unroll
    for (int ks = 0; ks < 4; ++ks)
      c = wmma16(a2[ks], load_b_nm(ps_w2_t, 128, tn * 16, ks * 32, lane), c);
    int n = tn * 16 + ln;
#pragma unroll
    for (int r = 0; r < 8; ++r) {
      int m = tm * 16 + half * 8 + r;
      int pg = pr0 + m;
      int typ = ptyp[(size_t)b * NP + pg];
      c[r] += pe_b2[n] + ps_b2[n] + pe_emb[(size_t)typ * ND + n];
    }
    acc[tn] = c;
  }
  layer_norm_wave(acc, pe_g, pe_lnb, ln);
  for (int tn = 0; tn < 8; ++tn) {
    int n = tn * 16 + ln;
#pragma unroll
    for (int r = 0; r < 8; ++r) {
      int m = tm * 16 + half * 8 + r;
      act[m * 136 + n] = (_Float16)acc[tn][r];
    }
  }
  // stage 3: hP = norm @ projP  (store f16 to global + keep in LDS for sigma head)
#pragma unroll
  for (int ks = 0; ks < 4; ++ks) a1[ks] = load_a_f16(act, 136, tm * 16, ks * 32, lane);
  _Float16* outp = hP16 + ((size_t)b * NP + pr0) * ND;
  for (int tn = 0; tn < 8; ++tn) {
    v8f c = {};
#pragma unroll
    for (int ks = 0; ks < 4; ++ks)
      c = wmma16(a1[ks], load_b_nm(projP_t, 128, tn * 16, ks * 32, lane), c);
    int n = tn * 16 + ln;
#pragma unroll
    for (int r = 0; r < 8; ++r) {
      int m = tm * 16 + half * 8 + r;
      _Float16 hv = (_Float16)c[r];
      outp[(size_t)m * ND + n] = hv;
      act[m * 136 + n] = hv;
    }
  }
  // stage 4: sigma head: g1 = gelu(hP @ sg_w1 + b1) ; row-dot with sg_w2 ; softplus
#pragma unroll
  for (int ks = 0; ks < 4; ++ks) a1[ks] = load_a_f16(act, 136, tm * 16, ks * 32, lane);
  v8f g1[8];
  for (int tn = 0; tn < 8; ++tn) {
    v8f c = {};
#pragma unroll
    for (int ks = 0; ks < 4; ++ks)
      c = wmma16(a1[ks], load_b_nm(sg_w1_t, 128, tn * 16, ks * 32, lane), c);
    int n = tn * 16 + ln;
#pragma unroll
    for (int r = 0; r < 8; ++r) c[r] = gelu_f(c[r] + sg_b1[n]);
    g1[tn] = c;
  }
  float w2v[8];
#pragma unroll
  for (int tn = 0; tn < 8; ++tn) w2v[tn] = sg_w2[tn * 16 + ln];
  for (int r = 0; r < 8; ++r) {
    float s = 0.0f;
#pragma unroll
    for (int tn = 0; tn < 8; ++tn) s += g1[tn][r] * w2v[tn];
    for (int off = 8; off >= 1; off >>= 1) s += __shfl_xor(s, off, 32);
    if (ln == 0) {
      int m = tm * 16 + half * 8 + r;
      size_t gp = (size_t)b * NP + pr0 + m;
      p_sigma[gp] = softplus_f(s + sg_b2[0]) + 0.001f + fmaxf(prad[gp], 0.0f);
    }
  }
}

// ---------------------------------------------------------------- logits + row stats
__global__ void __launch_bounds__(256) k_logits(
    const _Float16* __restrict__ hL16, const _Float16* __restrict__ hP16,
    const float* __restrict__ dustbin, const unsigned char* __restrict__ lmask,
    const unsigned char* __restrict__ pmask, float* __restrict__ maxw,
    float* __restrict__ rowent, float* __restrict__ lwtop, int* __restrict__ topj) {
  extern __shared__ char smem_raw[];
  float* lrow = (float*)smem_raw;  // [16][1032]
  const int b = blockIdx.x >> 3, lt = blockIdx.x & 7;
  const int t = threadIdx.x, lane = t & 31, wave = t >> 5;
  const int ln = lane & 15, half = lane >> 4;
  const _Float16* A = hL16 + ((size_t)b * NL + lt * 16) * ND;
  const _Float16* Bp = hP16 + (size_t)b * NP * ND;
  __builtin_prefetch(Bp, 0, 3);
  v16h af[4];
#pragma unroll
  for (int ks = 0; ks < 4; ++ks) af[ks] = load_a_f16(A, ND, 0, ks * 32, lane);
  for (int ct = wave; ct < 64; ct += 8) {
    v8f c = {};
#pragma unroll
    for (int ks = 0; ks < 4; ++ks)
      c = wmma16(af[ks], load_b_nm(Bp, ND, ct * 16, ks * 32, lane), c);
#pragma unroll
    for (int r = 0; r < 8; ++r)
      lrow[(half * 8 + r) * 1032 + ct * 16 + ln] = c[r] * SCALE_D;
  }
  __syncthreads();
  for (int rr = wave * 2; rr < wave * 2 + 2; ++rr) {
    const int l = lt * 16 + rr;
    const int lmv = lmask[(size_t)b * NL + l];
    // dustbin logit
    const _Float16* hrow = hL16 + ((size_t)b * NL + l) * ND;
    float dd = 0.0f;
    for (int d = lane; d < ND; d += 32) dd += (float)hrow[d] * dustbin[d];
    for (int off = 16; off >= 1; off >>= 1) dd += __shfl_xor(dd, off, 32);
    float dust = lmv ? dd * SCALE_D : NEGF;
    // online stats over the 1024 real logits
    float m = NEGF, S0 = 0.0f, S1 = 0.0f;
    float tv[4] = {NEGF, NEGF, NEGF, NEGF};
    int ti[4] = {-1, -1, -1, -1};
    for (int p = lane; p < NP; p += 32) {
      float x = lrow[rr * 1032 + p];
      if (!lmv || !pmask[(size_t)b * NP + p]) x = NEGF;
      float mn = fmaxf(m, x);
      float e0 = expf(m - mn), ex = expf(x - mn);
      S0 = S0 * e0 + ex;
      S1 = S1 * e0 + x * ex;
      m = mn;
      top4_insert(tv, ti, x, p);
    }
    // lane butterfly merge
    for (int off = 16; off >= 1; off >>= 1) {
      float m2 = __shfl_xor(m, off, 32);
      float s02 = __shfl_xor(S0, off, 32);
      float s12 = __shfl_xor(S1, off, 32);
      float mn = fmaxf(m, m2);
      float e1 = expf(m - mn), e2 = expf(m2 - mn);
      S0 = S0 * e1 + s02 * e2;
      S1 = S1 * e1 + s12 * e2;
      m = mn;
      float pv[4]; int pi[4];
#pragma unroll
      for (int j = 0; j < 4; ++j) {
        pv[j] = __shfl_xor(tv[j], off, 32);
        pi[j] = __shfl_xor(ti[j], off, 32);
      }
      for (int j = 0; j < 4; ++j) top4_insert(tv, ti, pv[j], pi[j]);
    }
    // fold in dustbin element
    {
      float mn = fmaxf(m, dust);
      float e1 = expf(m - mn), e2 = expf(dust - mn);
      S0 = S0 * e1 + e2;
      S1 = S1 * e1 + dust * e2;
      m = mn;
    }
    float lse = m + logf(S0);
    if (lane == 0) {
      size_t ro = (size_t)b * NL + l;
      maxw[ro] = expf(m - lse);            // max of softmax weights
      rowent[ro] = -(S1 / S0 - lse);       // entropy = -sum W logW
#pragma unroll
      for (int j = 0; j < 4; ++j) {
        topj[ro * 4 + j] = ti[j];
        lwtop[ro * 4 + j] = tv[j] - lse;
      }
    }
  }
}

// ---------------------------------------------------------------- final fuse
__global__ void __launch_bounds__(256) k_final(
    const float* __restrict__ lpos, const float* __restrict__ ppos,
    const int* __restrict__ aidx, const int* __restrict__ bidx,
    const unsigned char* __restrict__ lmask, const unsigned char* __restrict__ pmask,
    const _Float16* __restrict__ hL16, const _Float16* __restrict__ hP16,
    const float* __restrict__ p_sigma, const float* __restrict__ maxw,
    const float* __restrict__ rowent, const float* __restrict__ lwtop,
    const int* __restrict__ topj, const float* __restrict__ inv_w1,
    const float* __restrict__ inv_b1, const float* __restrict__ inv_w2,
    const float* __restrict__ inv_b2, const float* __restrict__ retrL,
    const float* __restrict__ retrP, float* __restrict__ out) {
  __shared__ float red[256];
  __shared__ float lg[128], pg[128], lzv[128], pzv[128];
  const int b = blockIdx.x, t = threadIdx.x;
  // sharp / neg-entropy / denomL
  float v0 = 0.0f, v1 = 0.0f, v2 = 0.0f;
  if (t < NL) {
    float lmv = lmask[(size_t)b * NL + t] ? 1.0f : 0.0f;
    v0 = maxw[(size_t)b * NL + t] * lmv;
    v1 = rowent[(size_t)b * NL + t] * lmv;
    v2 = lmv;
  }
  float sum_maxw = block_sum(v0, red);
  float sum_ent  = block_sum(v1, red);
  float sum_lm   = block_sum(v2, red);
  float denL = fmaxf(sum_lm, 1.0f);
  float sharp = sum_maxw / denL;
  float negent = -(sum_ent / denL);
  float vpm = 0.0f;
  for (int p = t; p < NP; p += 256) vpm += pmask[(size_t)b * NP + p] ? 1.0f : 0.0f;
  float denP = fmaxf(block_sum(vpm, red), 1.0f);
  // masked means of hL / hP
  if (t < ND) {
    float s = 0.0f;
    for (int l = 0; l < NL; ++l)
      s += (float)hL16[((size_t)b * NL + l) * ND + t] *
           (lmask[(size_t)b * NL + l] ? 1.0f : 0.0f);
    lg[t] = s / denL;
    float s2 = 0.0f;
    for (int p = 0; p < NP; ++p)
      s2 += (float)hP16[((size_t)b * NP + p) * ND + t] *
            (pmask[(size_t)b * NP + p] ? 1.0f : 0.0f);
    pg[t] = s2 / denP;
  }
  __syncthreads();
  // edge: K x K Gaussian log-mixture per sampled pair
  float lmix = 0.0f, vp = 0.0f;
  if (t < NM) {
    int ai = aidx[(size_t)b * NM + t];
    int bi = bidx[(size_t)b * NM + t];
    bool valid = lmask[(size_t)b * NL + ai] && lmask[(size_t)b * NL + bi] && (ai != bi);
    if (valid) {
      vp = 1.0f;
      const float* pa = lpos + ((size_t)b * NL + ai) * 3;
      const float* pb = lpos + ((size_t)b * NL + bi) * 3;
      float dx = pa[0] - pb[0], dy = pa[1] - pb[1], dz = pa[2] - pb[2];
      float dist = sqrtf(dx * dx + dy * dy + dz * dz);
      float ax[4], ay[4], az[4], sa[4], lwa[4];
      float bx[4], by[4], bz[4], sb[4], lwb[4];
#pragma unroll
      for (int k = 0; k < 4; ++k) {
        int ja = topj[((size_t)b * NL + ai) * 4 + k];
        int jb = topj[((size_t)b * NL + bi) * 4 + k];
        const float* qa = ppos + ((size_t)b * NP + ja) * 3;
        const float* qb = ppos + ((size_t)b * NP + jb) * 3;
        ax[k] = qa[0]; ay[k] = qa[1]; az[k] = qa[2];
        bx[k] = qb[0]; by[k] = qb[1]; bz[k] = qb[2];
        sa[k] = p_sigma[(size_t)b * NP + ja];
        sb[k] = p_sigma[(size_t)b * NP + jb];
        lwa[k] = lwtop[((size_t)b * NL + ai) * 4 + k];
        lwb[k] = lwtop[((size_t)b * NL + bi) * 4 + k];
      }
      float zz[16];
      float zm = NEGF;
      int c = 0;
      for (int i = 0; i < 4; ++i)
        for (int j = 0; j < 4; ++j) {
          float mx = ax[i] - bx[j], my = ay[i] - by[j], mz = az[i] - bz[j];
          float mu = sqrtf(mx * mx + my * my + mz * mz);
          float sg = sqrtf(sa[i] * sa[i] + sb[j] * sb[j] + 1e-6f);
          float dm = (dist - mu) / sg;
          float z = lwa[i] + lwb[j] - 0.5f * dm * dm - logf(sg) - HALF_LOG_2PI;
          zz[c++] = z;
          zm = fmaxf(zm, z);
        }
      float se = 0.0f;
      for (int i = 0; i < 16; ++i) se += expf(zz[i] - zm);
      lmix = zm + logf(se);
    }
  }
  float smix = block_sum(lmix, red);
  float svp  = block_sum(vp, red);
  float edge = tanhf((smix / fmaxf(svp, 1.0f)) * 0.2f);
  // tiny invariant MLP (3 -> 64 -> 1)
  if (t == 0) {
    float iv[3] = {sharp, negent, edge};
    float s = inv_b2[0];
    for (int j = 0; j < 64; ++j) {
      float h = inv_b1[j];
      for (int i = 0; i < 3; ++i) h += iv[i] * inv_w1[i * 64 + j];
      s += gelu_f(h) * inv_w2[j];
    }
    out[(size_t)b * 257] = s;
  }
  // retrieval mat-vecs + L2 normalize
  if (t < ND) {
    float s = 0.0f, s2 = 0.0f;
    for (int k = 0; k < ND; ++k) {
      s  += lg[k] * retrL[(size_t)k * ND + t];
      s2 += pg[k] * retrP[(size_t)k * ND + t];
    }
    lzv[t] = s; pzv[t] = s2;
  }
  __syncthreads();
  float nl = block_sum((t < ND) ? lzv[t] * lzv[t] : 0.0f, red);
  float np = block_sum((t < ND) ? pzv[t] * pzv[t] : 0.0f, red);
  if (t < ND) {
    out[(size_t)b * 257 + 1 + t]   = lzv[t] / fmaxf(sqrtf(nl), 1e-12f);
    out[(size_t)b * 257 + 129 + t] = pzv[t] / fmaxf(sqrtf(np), 1e-12f);
  }
}

// ---------------------------------------------------------------- launcher
extern "C" void kernel_launch(void* const* d_in, const int* in_sizes, int n_in,
                              void* d_out, int out_size, void* d_ws, size_t ws_size,
                              hipStream_t stream) {
  const float* l_pos   = (const float*)d_in[0];
  const float* l_x     = (const float*)d_in[1];
  const float* p_pos   = (const float*)d_in[2];
  const float* p_x     = (const float*)d_in[3];
  const float* p_score = (const float*)d_in[4];
  const float* p_rad   = (const float*)d_in[5];
  const int*   l_typ   = (const int*)d_in[6];
  const int*   p_typ   = (const int*)d_in[7];
  const int*   a_idx   = (const int*)d_in[8];
  const int*   b_idx   = (const int*)d_in[9];
  const unsigned char* l_mask = (const unsigned char*)d_in[10];
  const unsigned char* p_mask = (const unsigned char*)d_in[11];
  const float* le_emb = (const float*)d_in[12];
  const float* le_w1  = (const float*)d_in[13];
  const float* le_b1  = (const float*)d_in[14];
  const float* le_w2  = (const float*)d_in[15];
  const float* le_b2  = (const float*)d_in[16];
  const float* le_g   = (const float*)d_in[17];
  const float* le_lnb = (const float*)d_in[18];
  const float* pe_emb = (const float*)d_in[19];
  const float* pe_w1  = (const float*)d_in[20];
  const float* pe_b1  = (const float*)d_in[21];
  const float* pe_w2  = (const float*)d_in[22];
  const float* pe_b2  = (const float*)d_in[23];
  const float* ps_w1  = (const float*)d_in[24];
  const float* ps_b1  = (const float*)d_in[25];
  const float* ps_w2  = (const float*)d_in[26];
  const float* ps_b2  = (const float*)d_in[27];
  const float* pe_g   = (const float*)d_in[28];
  const float* pe_lnb = (const float*)d_in[29];
  const float* projL  = (const float*)d_in[30];
  const float* projP  = (const float*)d_in[31];
  const float* dustbin = (const float*)d_in[32];
  const float* sg_w1  = (const float*)d_in[33];
  const float* sg_b1  = (const float*)d_in[34];
  const float* sg_w2  = (const float*)d_in[35];
  const float* sg_b2  = (const float*)d_in[36];
  const float* inv_w1 = (const float*)d_in[37];
  const float* inv_b1 = (const float*)d_in[38];
  const float* inv_w2 = (const float*)d_in[39];
  const float* inv_b2 = (const float*)d_in[40];
  const float* retrL  = (const float*)d_in[41];
  const float* retrP  = (const float*)d_in[42];
  float* out = (float*)d_out;

  // workspace carve (256B aligned slices)
  char* w = (char*)d_ws;
  auto take = [&](size_t bytes) {
    char* p = w;
    w += (bytes + 255) & ~(size_t)255;
    return p;
  };
  _Float16* hL16    = (_Float16*)take((size_t)NB * NL * ND * 2);
  _Float16* hP16    = (_Float16*)take((size_t)NB * NP * ND * 2);
  float*    sigbuf  = (float*)take((size_t)NB * NP * 4);
  float*    maxwb   = (float*)take((size_t)NB * NL * 4);
  float*    rowentb = (float*)take((size_t)NB * NL * 4);
  float*    lwtopb  = (float*)take((size_t)NB * NL * 4 * 4);
  int*      topjb   = (int*)take((size_t)NB * NL * 4 * 4);
  _Float16* le_w2_t = (_Float16*)take(128 * 128 * 2);
  _Float16* projL_t = (_Float16*)take(128 * 128 * 2);
  _Float16* pe_w1_t = (_Float16*)take(192 * 128 * 2);
  _Float16* pe_w2_t = (_Float16*)take(128 * 128 * 2);
  _Float16* ps_w2_t = (_Float16*)take(128 * 128 * 2);
  _Float16* projP_t = (_Float16*)take(128 * 128 * 2);
  _Float16* sg_w1_t = (_Float16*)take(128 * 128 * 2);

  // weight conversions to f16 N-major
  k_cvt_nmajor<<<(128 * 128 + 255) / 256, 256, 0, stream>>>(le_w2, le_w2_t, 128, 128);
  k_cvt_nmajor<<<(128 * 128 + 255) / 256, 256, 0, stream>>>(projL, projL_t, 128, 128);
  k_cvt_nmajor<<<(192 * 128 + 255) / 256, 256, 0, stream>>>(pe_w1, pe_w1_t, 192, 128);
  k_cvt_nmajor<<<(128 * 128 + 255) / 256, 256, 0, stream>>>(pe_w2, pe_w2_t, 128, 128);
  k_cvt_nmajor<<<(128 * 128 + 255) / 256, 256, 0, stream>>>(ps_w2, ps_w2_t, 128, 128);
  k_cvt_nmajor<<<(128 * 128 + 255) / 256, 256, 0, stream>>>(projP, projP_t, 128, 128);
  k_cvt_nmajor<<<(128 * 128 + 255) / 256, 256, 0, stream>>>(sg_w1, sg_w1_t, 128, 128);

  k_embedL<<<NB, 256, 128 * 136 * 2, stream>>>(
      l_x, l_typ, le_emb, le_w1, le_b1, le_w2_t, le_b2, le_g, le_lnb, projL_t, hL16);

  k_embedP<<<NB * 8, 256, 2 * 128 * 136 * 2, stream>>>(
      p_x, p_score, p_rad, p_typ, pe_emb, pe_w1_t, pe_b1, pe_w2_t, pe_b2, ps_w1,
      ps_b1, ps_w2_t, ps_b2, pe_g, pe_lnb, projP_t, sg_w1_t, sg_b1, sg_w2, sg_b2,
      hP16, sigbuf);

  k_logits<<<NB * 8, 256, 16 * 1032 * 4, stream>>>(
      hL16, hP16, dustbin, l_mask, p_mask, maxwb, rowentb, lwtopb, topjb);

  k_final<<<NB, 256, 0, stream>>>(
      l_pos, p_pos, a_idx, b_idx, l_mask, p_mask, hL16, hP16, sigbuf, maxwb,
      rowentb, lwtopb, topjb, inv_w1, inv_b1, inv_w2, inv_b2, retrL, retrP, out);
}
